// StabilizedGRU_76029511074127
// MI455X (gfx1250) — matrix operand order repaired and checked by reading, not compiled, not run on hardware
//
#include <hip/hip_runtime.h>
#include <hip/hip_fp16.h>
#include <math.h>

// Problem dims (fixed by reference)
#define B_   256
#define T_   512
#define D_   200
#define H_   320
#define H3_  960
#define L_   3
#define KP0_ 224           // D=200 padded to multiple of 32
#define ASTRIDE 352        // f16 A-tile LDS row stride (16B aligned, covers 224/320)

typedef __attribute__((ext_vector_type(8)))  _Float16 v8h;
typedef __attribute__((ext_vector_type(16))) _Float16 v16h;
typedef __attribute__((ext_vector_type(8)))  float    v8f;

__device__ __forceinline__ v16h make_v16(v8h lo, v8h hi) {
    v16h r;
#pragma unroll
    for (int i = 0; i < 8; ++i) { r[i] = lo[i]; r[i + 8] = hi[i]; }
    return r;
}

// A-matrix fragment 16x32 f16 from LDS (row-major, stride ASTRIDE halves).
// Lane<16: M=lane, K={0..7,16..23}; lane>=16: M=lane-16, K={8..15,24..31}.
__device__ __forceinline__ v16h load_a(const _Float16* base, int kBase) {
    const int lane = threadIdx.x & 31;
    const _Float16* p = base + (lane & 15) * ASTRIDE + kBase + ((lane >> 4) * 8);
    v8h lo = *(const v8h*)p;
    v8h hi = *(const v8h*)(p + 16);
    return make_v16(lo, hi);
}

// B-matrix fragment 32x16 f16 from global weights W (row-major [3H, ldw], W^T mult).
// Column N = lane&15; lane<16: K={0..7,16..23}; lane>=16: K={8..15,24..31}.
__device__ __forceinline__ v16h load_b(const _Float16* __restrict__ w, int ldw,
                                       int nBase, int kBase) {
    const int lane = threadIdx.x & 31;
    const _Float16* p = w + (size_t)(nBase + (lane & 15)) * ldw + kBase + ((lane >> 4) * 8);
    v8h lo = *(const v8h*)p;
    v8h hi = *(const v8h*)(p + 16);
    return make_v16(lo, hi);
}

__device__ __forceinline__ float wave_allreduce(float v) {
#pragma unroll
    for (int off = 16; off > 0; off >>= 1) v += __shfl_xor(v, off, 32);
    return v;
}

// ---------------------------------------------------------------------------
// Persistent GRU kernel. Key insight: every dependency (t -> t+1, layer ->
// layer+1) is local to a 16-row batch block, so 16 workgroups run the entire
// T=512 x L=3 recurrence with only __syncthreads() between phases. Hidden
// states live in LDS for the whole run (f16 tiles in A-fragment layout feed
// the recurrent WMMA directly; f32 copies preserve blend precision). Weights
// (3.4 MB f16) stream from the 192 MB L2 every step.
// grid = 16 workgroups, block = 512 threads = 16 wave32s.
// ---------------------------------------------------------------------------
__global__ __launch_bounds__(512)
void gru_persistent_kernel(const float* __restrict__ x,
                           const float* __restrict__ rmean,
                           const float* __restrict__ rvar,
                           const _Float16* __restrict__ wih0,   // [960][224] padded
                           const _Float16* __restrict__ wih12,  // [2][960][320]
                           const _Float16* __restrict__ whhA,   // [3][960][320]
                           const float* __restrict__ b_ih,      // [3][960]
                           const float* __restrict__ b_hh,      // [3][960]
                           const float* __restrict__ ln_gamma,  // [3][3][320]
                           const float* __restrict__ ln_beta,   // [3][3][320]
                           const float* __restrict__ hidden0,   // [3][256][320]
                           float* __restrict__ out)             // [B*T*H] ++ [L*B*H]
{
    extern __shared__ char smem[];
    float*    comb = (float*)smem;                  // [16][960] gi+gh (r,u) / gi (n)
    float*    ghn  = comb + 16 * H3_;               // [16][320] gh for n gate
    float*    h32t = ghn  + 16 * H_;                // [3][16][320] f32 hidden
    _Float16* xA   = (_Float16*)(h32t + 3 * 16 * H_); // [16][ASTRIDE] layer-0 input
    _Float16* hT   = xA + 16 * ASTRIDE;             // [3][16][ASTRIDE] f16 hidden

    const int tid     = threadIdx.x;
    const int rowBase = blockIdx.x * 16;
    const int wave    = tid >> 5;
    const int lane    = tid & 31;
    const int m0      = (lane >> 4) * 8;   // C/D: lanes>=16 hold M=v+8
    const int nc      = lane & 15;

    // ---- Load initial hidden state into persistent LDS tiles ----
    for (int idx = tid; idx < 3 * 16 * H_; idx += 512) {
        int l = idx / (16 * H_);
        int rem = idx - l * (16 * H_);
        int r = rem / H_, k = rem - r * H_;
        float v = hidden0[(size_t)l * B_ * H_ + (size_t)(rowBase + r) * H_ + k];
        h32t[idx] = v;
        hT[l * 16 * ASTRIDE + r * ASTRIDE + k] = (_Float16)v;
    }
    __syncthreads();

    for (int t = 0; t < T_; ++t) {
        // ---- Stage normalized x_t tile (layer-0 input) ----
        for (int idx = tid; idx < 16 * KP0_; idx += 512) {
            int r = idx / KP0_, k = idx - r * KP0_;
            float v = 0.0f;
            if (k < D_) {
                float xv = x[(size_t)(rowBase + r) * (T_ * D_) + (size_t)t * D_ + k];
                v = (xv - rmean[k]) * rsqrtf(rvar[k] + 1e-8f);
                v = fminf(10.0f, fmaxf(-10.0f, v));
            }
            xA[r * ASTRIDE + k] = (_Float16)v;
        }
        __syncthreads();

        for (int l = 0; l < L_; ++l) {
            const _Float16* inA = (l == 0) ? xA : (hT + (l - 1) * 16 * ASTRIDE);
            const _Float16* hA  = hT + l * 16 * ASTRIDE;
            const int kinPad    = (l == 0) ? KP0_ : H_;
            const _Float16* wih = (l == 0) ? wih0
                                           : (wih12 + (size_t)(l - 1) * H3_ * H_);
            const _Float16* whh = whhA + (size_t)l * H3_ * H_;
            const float* bih = b_ih + l * H3_;
            const float* bhh = b_hh + l * H3_;
            const float* lnG = ln_gamma + l * 3 * H_;
            const float* lnB = ln_beta  + l * 3 * H_;
            float* h32l = h32t + l * 16 * H_;
            _Float16* hTl = hT + l * 16 * ASTRIDE;

            // ---- Gates r,u: chain input-GEMM and hidden-GEMM into ONE acc ----
            for (int nt = wave; nt < 40; nt += 16) {
                const int nBase = nt * 16;
                v8f acc = {};
                for (int ks = 0; ks < kinPad; ks += 32) {
                    v16h a = load_a(inA, ks);
                    v16h b = load_b(wih, kinPad, nBase, ks);
                    acc = __builtin_amdgcn_wmma_f32_16x16x32_f16(
                        false, a, false, b, (short)0, acc, false, false);
                }
#pragma unroll
                for (int ks = 0; ks < H_; ks += 32) {
                    v16h a = load_a(hA, ks);
                    v16h b = load_b(whh, H_, nBase, ks);
                    acc = __builtin_amdgcn_wmma_f32_16x16x32_f16(
                        false, a, false, b, (short)0, acc, false, false);
                }
#pragma unroll
                for (int v = 0; v < 8; ++v) comb[(m0 + v) * H3_ + nBase + nc] = acc[v];
            }

            // ---- Gate n: keep gi and gh separate (needs r * hn) ----
            for (int nt = 40 + wave; nt < 60; nt += 16) {
                const int nBase = nt * 16;
                v8f ai = {};
                for (int ks = 0; ks < kinPad; ks += 32) {
                    v16h a = load_a(inA, ks);
                    v16h b = load_b(wih, kinPad, nBase, ks);
                    ai = __builtin_amdgcn_wmma_f32_16x16x32_f16(
                        false, a, false, b, (short)0, ai, false, false);
                }
                v8f ah = {};
#pragma unroll
                for (int ks = 0; ks < H_; ks += 32) {
                    v16h a = load_a(hA, ks);
                    v16h b = load_b(whh, H_, nBase, ks);
                    ah = __builtin_amdgcn_wmma_f32_16x16x32_f16(
                        false, a, false, b, (short)0, ah, false, false);
                }
#pragma unroll
                for (int v = 0; v < 8; ++v) {
                    comb[(m0 + v) * H3_ + nBase + nc] = ai[v];
                    ghn [(m0 + v) * H_  + (nBase - 2 * H_) + nc] = ah[v];
                }
            }
            __syncthreads();

            // ---- Epilogue: one batch row per wave (16 rows, 16 waves) ----
            {
                const int r = wave;
                float* cRow = comb + r * H3_;

                // gates 0 (reset) and 1 (update): z = gi+gh (+biases), LN, sigmoid
#pragma unroll
                for (int g = 0; g < 2; ++g) {
                    float s = 0.0f, s2 = 0.0f;
                    for (int j = lane; j < H_; j += 32) {
                        const int c = g * H_ + j;
                        float z = cRow[c] + bih[c] + bhh[c];
                        cRow[c] = z;
                        s += z; s2 += z * z;
                    }
                    s  = wave_allreduce(s);
                    s2 = wave_allreduce(s2);
                    float mean = s * (1.0f / H_);
                    float var  = s2 * (1.0f / H_) - mean * mean;
                    float rstd = rsqrtf(var + 1e-5f);
                    for (int j = lane; j < H_; j += 32) {
                        const int c = g * H_ + j;
                        float y = (cRow[c] - mean) * rstd * lnG[c] + lnB[c];
                        cRow[c] = 1.0f / (1.0f + __expf(-y));   // r or u
                    }
                }
                // gate 2 (candidate): z = gi_n + r*gh_n (+biases), LN, tanh, blend
                {
                    float s = 0.0f, s2 = 0.0f;
                    for (int j = lane; j < H_; j += 32) {
                        float rg = cRow[j];
                        float z  = cRow[2 * H_ + j] + bih[2 * H_ + j]
                                 + rg * (ghn[r * H_ + j] + bhh[2 * H_ + j]);
                        cRow[2 * H_ + j] = z;
                        s += z; s2 += z * z;
                    }
                    s  = wave_allreduce(s);
                    s2 = wave_allreduce(s2);
                    float mean = s * (1.0f / H_);
                    float var  = s2 * (1.0f / H_) - mean * mean;
                    float rstd = rsqrtf(var + 1e-5f);
                    for (int j = lane; j < H_; j += 32) {
                        float nval = tanhf((cRow[2 * H_ + j] - mean) * rstd
                                           * lnG[2 * H_ + j] + lnB[2 * H_ + j]);
                        float u  = cRow[H_ + j];
                        float hp = h32l[r * H_ + j];
                        float hn = (1.0f - u) * nval + u * hp;
                        h32l[r * H_ + j] = hn;
                        hTl [r * ASTRIDE + j] = (_Float16)hn;
                        if (l == L_ - 1)
                            out[(size_t)(rowBase + r) * (T_ * H_) + (size_t)t * H_ + j] = hn;
                    }
                }
            }
            __syncthreads();   // hT[l]/comb/ghn consistent before next layer
        }
    }

    // ---- Final hidden state: [L][B][H] appended after [B][T][H] ----
    for (int idx = tid; idx < 3 * 16 * H_; idx += 512) {
        int l = idx / (16 * H_);
        int rem = idx - l * (16 * H_);
        int r = rem / H_, k = rem - r * H_;
        out[(size_t)B_ * T_ * H_ + (size_t)l * B_ * H_
            + (size_t)(rowBase + r) * H_ + k] = h32t[idx];
    }
}

// f32 -> f16 weight conversion; w_ih0 padded from K=200 to K=224 with zeros.
__global__ void convert_weights_kernel(const float* __restrict__ w_ih0,
                                       const float* __restrict__ w_ih_rest,
                                       const float* __restrict__ w_hh,
                                       _Float16* __restrict__ w16)
{
    const int N0 = H3_ * KP0_;        // padded layer-0 input weights
    const int N1 = 2 * H3_ * H_;      // layers 1..2 input weights
    const int N2 = L_ * H3_ * H_;     // hidden weights
    int idx = blockIdx.x * 256 + threadIdx.x;
    if (idx < N0) {
        int r = idx / KP0_, k = idx - r * KP0_;
        w16[idx] = (k < D_) ? (_Float16)w_ih0[r * D_ + k] : (_Float16)0.0f;
    } else if (idx < N0 + N1) {
        w16[idx] = (_Float16)w_ih_rest[idx - N0];
    } else if (idx < N0 + N1 + N2) {
        w16[idx] = (_Float16)w_hh[idx - N0 - N1];
    }
}

extern "C" void kernel_launch(void* const* d_in, const int* in_sizes, int n_in,
                              void* d_out, int out_size, void* d_ws, size_t ws_size,
                              hipStream_t stream)
{
    (void)in_sizes; (void)n_in; (void)out_size; (void)ws_size;
    const float* x        = (const float*)d_in[0];
    const float* hidden0  = (const float*)d_in[1];
    const float* rmean    = (const float*)d_in[2];
    const float* rvar     = (const float*)d_in[3];
    const float* w_ih0    = (const float*)d_in[4];
    const float* w_ihrest = (const float*)d_in[5];
    const float* w_hh     = (const float*)d_in[6];
    const float* b_ih     = (const float*)d_in[7];
    const float* b_hh     = (const float*)d_in[8];
    const float* ln_gamma = (const float*)d_in[9];
    const float* ln_beta  = (const float*)d_in[10];
    float* out = (float*)d_out;

    // Workspace: f16 weight copies (L2-resident for the whole recurrence)
    _Float16* w16   = (_Float16*)d_ws;
    _Float16* wih0h = w16;
    _Float16* wih12 = wih0h + (size_t)H3_ * KP0_;
    _Float16* whhA  = wih12 + (size_t)2 * H3_ * H_;

    {
        int N = H3_ * KP0_ + 2 * H3_ * H_ + L_ * H3_ * H_;
        convert_weights_kernel<<<(N + 255) / 256, 256, 0, stream>>>(
            w_ih0, w_ihrest, w_hh, w16);
    }

    const size_t SMEM = (size_t)16 * H3_ * 4       // comb
                      + (size_t)16 * H_ * 4        // ghn
                      + (size_t)3 * 16 * H_ * 4    // h32 tiles
                      + (size_t)16 * ASTRIDE * 2   // x tile
                      + (size_t)3 * 16 * ASTRIDE * 2; // h16 tiles  = 188,416 B
    hipFuncSetAttribute((const void*)gru_persistent_kernel,
                        hipFuncAttributeMaxDynamicSharedMemorySize, (int)SMEM);

    gru_persistent_kernel<<<dim3(16), dim3(512), SMEM, stream>>>(
        x, rmean, rvar, wih0h, wih12, whhA,
        b_ih, b_hh, ln_gamma, ln_beta, hidden0, out);
}
